// UNet_3118146257345
// MI455X (gfx1250) — matrix operand order repaired and checked by reading, not compiled, not run on hardware
//
#include <hip/hip_runtime.h>
#include <hip/hip_bf16.h>

// ============================================================================
// Point-cloud UNet forward for MI455X (gfx1250).
// Dense work -> one strided-batched GEMM on V_WMMA_F32_16X16X32_BF16:
//   * 256 threads = 8 wave32 waves, 64x64 block tile, each wave 16x32 (2 WMMA)
//   * A tile staged as packed-bf16 pairs in LDS (fragment = 2x ds_load_b128)
//   * X tile staged in fp32 LDS; interior tiles use GLOBAL_LOAD_ASYNC_TO_LDS_B32
//     (ASYNCcnt + s_wait_asynccnt), edge/strided tiles use branchless VGPR path
// Irregular ops (FPS, KNN, gathers, softmax, aggregation, tconv) are
// dedicated lightweight kernels.
// ============================================================================

typedef __attribute__((ext_vector_type(16))) __bf16 v16bf;
typedef __attribute__((ext_vector_type(8)))  float  v8f;
typedef __attribute__((address_space(3)))    float  lds_f32;

// ---------------------------------------------------------------------------
// Generic strided batched GEMM:
//   Y[b, m, n] = act( scale * sum_k A[b*aB + m*aM + k*aK] * X[b*xB + k*xK + n*xN]
//                     + bias[m] + res[b*rB + m*rM + n] )
// ---------------------------------------------------------------------------
__global__ __launch_bounds__(256) void wmma_gemm_kernel(
    const float* __restrict__ Am, long aB, long aM, long aK,
    const float* __restrict__ Xm, long xB, long xK, long xN,
    const float* __restrict__ bias,
    const float* __restrict__ res, long rB, long rM,
    float* __restrict__ Y, long yB, long yM,
    int M, int N, int K, float scale, int relu)
{
  __shared__ unsigned sWp[64][16];     // A tile, bf16 pairs: [m][k/2], 64B rows
  __shared__ float    sXf[32][65];     // X tile, fp32: [k][n], padded
  const int tid  = threadIdx.x;
  const int b    = blockIdx.z;
  const int m0   = blockIdx.y * 64;
  const int n0   = blockIdx.x * 64;
  const int wave = tid >> 5, lane = tid & 31;
  const int wm = wave >> 1, wn = wave & 1;      // 4x2 wave grid
  const int lhi = lane >> 4, llo = lane & 15;

  v8f acc0 = {0.f,0.f,0.f,0.f,0.f,0.f,0.f,0.f};
  v8f acc1 = {0.f,0.f,0.f,0.f,0.f,0.f,0.f,0.f};

  for (int k0 = 0; k0 < K; k0 += 32) {
    // ---- stage A (64 x 32) as packed bf16 pairs, branchless ----
#pragma unroll
    for (int i = 0; i < 4; ++i) {
      int t  = tid + i * 256;               // 0..1023
      int mm = t >> 4, kp = t & 15;
      int mg = m0 + mm, kg = k0 + 2 * kp;
      long base = (long)b * aB + (long)(mg < M ? mg : M - 1) * aM;
      float f0 = Am[base + (long)(kg     < K ? kg     : K - 1) * aK];
      float f1 = Am[base + (long)(kg + 1 < K ? kg + 1 : K - 1) * aK];
      if (!(mg < M && kg     < K)) f0 = 0.f;
      if (!(mg < M && kg + 1 < K)) f1 = 0.f;
      union { __bf16 h[2]; unsigned u; } pk;
      pk.h[0] = (__bf16)f0; pk.h[1] = (__bf16)f1;
      sWp[mm][kp] = pk.u;
    }
    // ---- stage X (32 x 64) fp32 ----
    bool interior = (xN == 1) && (k0 + 32 <= K) && (n0 + 64 <= N);
    if (interior) {
      // async DMA straight into LDS (no VGPR round trip), per-lane b32
#pragma unroll
      for (int i = 0; i < 8; ++i) {
        int t  = tid + i * 256;             // 0..2047
        int kk = t >> 6, nn = t & 63;
        unsigned ldsoff =
            (unsigned)(unsigned long long)(lds_f32*)&sXf[kk][nn];
        const float* g = Xm + (long)b * xB + (long)(k0 + kk) * xK + (n0 + nn);
        asm volatile("global_load_async_to_lds_b32 %0, %1, off"
                     :: "v"(ldsoff), "v"(g) : "memory");
      }
      asm volatile("s_wait_asynccnt 0x0" ::: "memory");
    } else {
#pragma unroll
      for (int i = 0; i < 8; ++i) {
        int t  = tid + i * 256;
        int kk = t >> 6, nn = t & 63;
        int kg = k0 + kk, ng = n0 + nn;
        long off = (long)b * xB + (long)(kg < K ? kg : K - 1) * xK
                 + (long)(ng < N ? ng : N - 1) * xN;
        float f = Xm[off];
        sXf[kk][nn] = (kg < K && ng < N) ? f : 0.f;
      }
    }
    if (k0 + 32 < K && tid == 0)
      __builtin_prefetch(Xm + (long)b * xB + (long)(k0 + 32) * xK + (long)n0 * xN, 0, 1);
    __syncthreads();

    // ---- A fragment (ISA 7.12.2 16-bit A 16x32): two aligned 16B LDS reads
    union { unsigned u[8]; v16bf v; } ua;
    {
      const unsigned* r32 = &sWp[wm * 16 + llo][0];
      uint4 lo = *(const uint4*)&r32[lhi * 4];
      uint4 hi = *(const uint4*)&r32[8 + lhi * 4];
      ua.u[0] = lo.x; ua.u[1] = lo.y; ua.u[2] = lo.z; ua.u[3] = lo.w;
      ua.u[4] = hi.x; ua.u[5] = hi.y; ua.u[6] = hi.z; ua.u[7] = hi.w;
    }
    // ---- B fragments: fp32 LDS -> packed bf16 pairs in registers
    union { unsigned u[8]; v16bf v; } ub0, ub1;
    const int col0 = wn * 32 + llo, col1 = col0 + 16;
#pragma unroll
    for (int h2 = 0; h2 < 8; ++h2) {
      int kr = lhi * 8 + 2 * h2 + (h2 >= 4 ? 8 : 0);
      union { __bf16 h[2]; unsigned u; } p0, p1;
      p0.h[0] = (__bf16)sXf[kr][col0]; p0.h[1] = (__bf16)sXf[kr + 1][col0];
      p1.h[0] = (__bf16)sXf[kr][col1]; p1.h[1] = (__bf16)sXf[kr + 1][col1];
      ub0.u[h2] = p0.u; ub1.u[h2] = p1.u;
    }

    acc0 = __builtin_amdgcn_wmma_f32_16x16x32_bf16(false, ua.v, false, ub0.v,
                                                   (short)0, acc0, false, false);
    acc1 = __builtin_amdgcn_wmma_f32_16x16x32_bf16(false, ua.v, false, ub1.v,
                                                   (short)0, acc1, false, false);
    __syncthreads();
  }

  // C/D layout: vgpr q -> row q + 8*lhi, col llo (per 16x16 tile)
#pragma unroll
  for (int q = 0; q < 8; ++q) {
    int mg  = m0 + wm * 16 + q + 8 * lhi;
    int ng0 = n0 + wn * 32 + llo;
    int ng1 = ng0 + 16;
    if (mg < M) {
      float bv = bias ? bias[mg] : 0.f;
      if (ng0 < N) {
        float v = acc0[q] * scale + bv;
        if (res) v += res[(long)b * rB + (long)mg * rM + ng0];
        if (relu && v < 0.f) v = 0.f;
        Y[(long)b * yB + (long)mg * yM + ng0] = v;
      }
      if (ng1 < N) {
        float v = acc1[q] * scale + bv;
        if (res) v += res[(long)b * rB + (long)mg * rM + ng1];
        if (relu && v < 0.f) v = 0.f;
        Y[(long)b * yB + (long)mg * yM + ng1] = v;
      }
    }
  }
}

// ---------------------------------------------------------------------------
// Farthest point sampling: one block per batch, xyz (B,3,N) -> out (B,npoint)
// ---------------------------------------------------------------------------
__global__ __launch_bounds__(256) void fps_kernel(
    const float* __restrict__ xyz, int* __restrict__ out,
    float* __restrict__ dist, int N, int npoint)
{
  const int b = blockIdx.x, tid = threadIdx.x;
  __shared__ float rd[256];
  __shared__ int   ri[256];
  __shared__ int   sfar;
  __shared__ float c[3];
  for (int n = tid; n < N; n += 256) dist[(long)b * N + n] = 1e10f;
  if (tid == 0) sfar = 0;
  __syncthreads();
  for (int s = 0; s < npoint; ++s) {
    if (tid == 0) {
      int far = sfar;
      out[(long)b * npoint + s] = far;
      c[0] = xyz[(long)b * 3 * N + far];
      c[1] = xyz[(long)b * 3 * N + N + far];
      c[2] = xyz[(long)b * 3 * N + 2 * N + far];
    }
    __syncthreads();
    float lm = -1.f; int li = 0;
    for (int n = tid; n < N; n += 256) {
      float dx = xyz[(long)b * 3 * N + n]         - c[0];
      float dy = xyz[(long)b * 3 * N + N + n]     - c[1];
      float dz = xyz[(long)b * 3 * N + 2 * N + n] - c[2];
      float d = dx * dx + dy * dy + dz * dz;
      float old = dist[(long)b * N + n];
      if (d < old) { old = d; dist[(long)b * N + n] = d; }
      if (old > lm) { lm = old; li = n; }
    }
    rd[tid] = lm; ri[tid] = li;
    __syncthreads();
    for (int off = 128; off > 0; off >>= 1) {
      if (tid < off && rd[tid + off] > rd[tid]) { rd[tid] = rd[tid + off]; ri[tid] = ri[tid + off]; }
      __syncthreads();
    }
    if (tid == 0) sfar = ri[0];
    __syncthreads();
  }
}

// ---------------------------------------------------------------------------
// KNN (k=16): q (B,3,M) vs r (B,3,N) -> idx (B,M,16). One wave per query.
// ---------------------------------------------------------------------------
__global__ __launch_bounds__(256) void knn_kernel(
    const float* __restrict__ q, const float* __restrict__ r,
    int* __restrict__ out, int B, int M, int N)
{
  __shared__ float sd[8][32][16];
  __shared__ int   si[8][32][16];
  const int wave = threadIdx.x >> 5, lane = threadIdx.x & 31;
  long g = (long)blockIdx.x * 8 + wave;
  if (g >= (long)B * M) return;   // grid sized so this never fires
  int b = (int)(g / M), m = (int)(g % M);
  float qx = q[(long)b * 3 * M + m];
  float qy = q[(long)b * 3 * M + M + m];
  float qz = q[(long)b * 3 * M + 2 * M + m];
#pragma unroll
  for (int j = 0; j < 16; ++j) { sd[wave][lane][j] = 1e30f; si[wave][lane][j] = 0; }
  for (int n = lane; n < N; n += 32) {
    float dx = r[(long)b * 3 * N + n]         - qx;
    float dy = r[(long)b * 3 * N + N + n]     - qy;
    float dz = r[(long)b * 3 * N + 2 * N + n] - qz;
    float d = dx * dx + dy * dy + dz * dz;
    if (d < sd[wave][lane][15]) {
      int p = 15;
      while (p > 0 && sd[wave][lane][p - 1] > d) {
        sd[wave][lane][p] = sd[wave][lane][p - 1];
        si[wave][lane][p] = si[wave][lane][p - 1];
        --p;
      }
      sd[wave][lane][p] = d; si[wave][lane][p] = n;
    }
  }
  __syncthreads();
  if (lane == 0) {
    for (int t = 0; t < 16; ++t) {
      float best = 1e30f; int bi = 0, bl = 0, bp = 0;
      for (int l2 = 0; l2 < 32; ++l2)
        for (int p = 0; p < 16; ++p) {
          float d = sd[wave][l2][p];
          if (d < best) { best = d; bi = si[wave][l2][p]; bl = l2; bp = p; }
        }
      sd[wave][bl][bp] = 1e30f;
      out[((long)b * M + m) * 16 + t] = bi;
    }
  }
}

// ---------------------------------------------------------------------------
// Small elementwise / gather kernels
// ---------------------------------------------------------------------------
__global__ void gather_cn_kernel(float* __restrict__ out, const float* __restrict__ src,
                                 const int* __restrict__ idx, int B, int C, int Ns, int Md)
{
  long i = (long)blockIdx.x * 256 + threadIdx.x;
  if (i >= (long)B * C * Md) return;
  int m = (int)(i % Md); long t = i / Md; int c = (int)(t % C); int b = (int)(t / C);
  out[i] = src[((long)b * C + c) * Ns + idx[(long)b * Md + m]];
}

// g (B, 3+C, M, K): first 3 ch = gathered xyz - new_xyz, rest = gathered points
__global__ void group_kernel(float* __restrict__ g, const float* __restrict__ xyz,
                             const float* __restrict__ nxyz, const float* __restrict__ pts,
                             const int* __restrict__ idx, int B, int C, int N, int M, int K)
{
  long i = (long)blockIdx.x * 256 + threadIdx.x;
  int CT = 3 + C;
  if (i >= (long)B * CT * M * K) return;
  int k = (int)(i % K); long t = i / K; int m = (int)(t % M); t /= M;
  int c = (int)(t % CT); int b = (int)(t / CT);
  int id = idx[((long)b * M + m) * K + k];
  float v;
  if (c < 3) v = xyz[((long)b * 3 + c) * N + id] - nxyz[((long)b * 3 + c) * M + m];
  else       v = pts[((long)b * C + (c - 3)) * N + id];
  g[i] = v;
}

// out[b,c,n,k] = a[b,c,n] - src[b,c,idx[b,n,k]]
__global__ void bsg_kernel(float* __restrict__ out, const float* __restrict__ a,
                           const float* __restrict__ src, const int* __restrict__ idx,
                           int B, int C, int N, int K)
{
  long i = (long)blockIdx.x * 256 + threadIdx.x;
  if (i >= (long)B * C * N * K) return;
  int k = (int)(i % K); long t = i / K; int n = (int)(t % N); t /= N;
  int c = (int)(t % C); int b = (int)(t / C);
  int id = idx[((long)b * N + n) * K + k];
  out[i] = a[((long)b * C + c) * N + n] - src[((long)b * C + c) * N + id];
}

__global__ void add_inplace_kernel(float* __restrict__ a, const float* __restrict__ b, long n)
{
  long i = (long)blockIdx.x * 256 + threadIdx.x;
  if (i < n) a[i] += b[i];
}

__global__ void avg2_kernel(float* __restrict__ o, const float* __restrict__ a,
                            const float* __restrict__ b, long n)
{
  long i = (long)blockIdx.x * 256 + threadIdx.x;
  if (i < n) o[i] = 0.5f * (a[i] + b[i]);
}

__global__ void softmax_last_kernel(float* __restrict__ x, long rows, int L)
{
  long r = (long)blockIdx.x * 256 + threadIdx.x;
  if (r >= rows) return;
  float* p = x + r * L;
  float mx = -1e30f;
  for (int l = 0; l < L; ++l) mx = p[l] > mx ? p[l] : mx;
  float s = 0.f;
  for (int l = 0; l < L; ++l) { float e = __expf(p[l] - mx); p[l] = e; s += e; }
  float inv = 1.f / s;
  for (int l = 0; l < L; ++l) p[l] *= inv;
}

// out[b,c,n] = sum_k attn[b,c,n,k] * (v[b,c,n] + pe[b,c,n,k])
__global__ void attn_agg_kernel(float* __restrict__ out, const float* __restrict__ attn,
                                const float* __restrict__ v, const float* __restrict__ pe,
                                int B, int C, int N, int K)
{
  long i = (long)blockIdx.x * 256 + threadIdx.x;
  if (i >= (long)B * C * N) return;
  float vb = v[i];
  const float* ap = attn + i * K;
  const float* pp = pe + i * K;
  float s = 0.f;
  for (int k = 0; k < K; ++k) s += ap[k] * (vb + pp[k]);
  out[i] = s;
}

__global__ void maxpool_last_kernel(float* __restrict__ out, const float* __restrict__ in,
                                    long rows, int L)
{
  long r = (long)blockIdx.x * 256 + threadIdx.x;
  if (r >= rows) return;
  const float* p = in + r * L;
  float m = p[0];
  for (int l = 1; l < L; ++l) m = p[l] > m ? p[l] : m;
  out[r] = m;
}

// transposed conv1d, stride==Kp: y[b,o,l*Kp+kp] = bias[o] + sum_c x[b,c,l]*w[c,o,kp]
__global__ void tconv_kernel(const float* __restrict__ x, const float* __restrict__ w,
                             const float* __restrict__ bias, float* __restrict__ y,
                             int B, int C, int O, int L, int Kp)
{
  long i = (long)blockIdx.x * 256 + threadIdx.x;
  if (i >= (long)B * O * L * Kp) return;
  int kp = (int)(i % Kp); long t = i / Kp; int l = (int)(t % L); t /= L;
  int o = (int)(t % O); int b = (int)(t / O);
  float s = bias[o];
  for (int c = 0; c < C; ++c)
    s += x[((long)b * C + c) * L + l] * w[((long)c * O + o) * Kp + kp];
  y[((long)b * O + o) * (L * Kp) + l * Kp + kp] = s;
}

// out (B, C1+C2, N) = concat(A (B,C1,N), Bp (B,C2,N or broadcast N=1), dim=1)
__global__ void chan_cat_kernel(float* __restrict__ out, const float* __restrict__ A, int C1,
                                const float* __restrict__ Bp, int C2, int N, int bcast, int B)
{
  long i = (long)blockIdx.x * 256 + threadIdx.x;
  int CT = C1 + C2;
  if (i >= (long)B * CT * N) return;
  int n = (int)(i % N); long t = i / N; int c = (int)(t % CT); int b = (int)(t / CT);
  out[i] = (c < C1) ? A[((long)b * C1 + c) * N + n]
                    : (bcast ? Bp[(long)b * C2 + (c - C1)]
                             : Bp[((long)b * C2 + (c - C1)) * N + n]);
}

// dst[row*Ldst + off + j] = src[row*Lsrc + j]
__global__ void copy_into_kernel(float* __restrict__ dst, int Ldst, int off,
                                 const float* __restrict__ src, int Lsrc, long rows)
{
  long i = (long)blockIdx.x * 256 + threadIdx.x;
  if (i >= rows * Lsrc) return;
  long r = i / Lsrc; int j = (int)(i % Lsrc);
  dst[r * Ldst + off + j] = src[i];
}

// fm[c,d] = W[c,d] / ||W[:,d]||
__global__ void colnorm_kernel(const float* __restrict__ W, float* __restrict__ fm,
                               int dim, int nd)
{
  int d = blockIdx.x * 256 + threadIdx.x;
  if (d >= nd) return;
  float s = 0.f;
  for (int c = 0; c < dim; ++c) { float v = W[(long)c * nd + d]; s += v * v; }
  float inv = rsqrtf(s);
  for (int c = 0; c < dim; ++c) fm[(long)c * nd + d] = W[(long)c * nd + d] * inv;
}

// ============================================================================
// Host orchestration
// ============================================================================
static inline dim3 grid1(long total) { return dim3((unsigned)((total + 255) / 256)); }

static void gemm(hipStream_t st, int Bc, int M, int N, int K,
                 const float* A, long aB, long aM, long aK,
                 const float* X, long xB, long xK, long xN,
                 const float* bias, float* Y, long yB, long yM,
                 float scale = 1.f, int relu = 0,
                 const float* res = nullptr, long rB = 0, long rM = 0)
{
  dim3 g((N + 63) / 64, (M + 63) / 64, Bc);
  wmma_gemm_kernel<<<g, 256, 0, st>>>(A, aB, aM, aK, X, xB, xK, xN,
                                      bias, res, rB, rM, Y, yB, yM, M, N, K, scale, relu);
}

extern "C" void kernel_launch(void* const* d_in, const int* in_sizes, int n_in,
                              void* d_out, int out_size, void* d_ws, size_t ws_size,
                              hipStream_t stream)
{
  (void)in_sizes; (void)n_in; (void)out_size; (void)ws_size;
  const int B = 8;
  hipStream_t st = stream;

  int ii = 0;
  auto F = [&]() { return (const float*)d_in[ii++]; };

  const float* pc = F();                                           // (B,3,2048)
  const float *sa1w0 = F(), *sa1b0 = F(), *sa1w1 = F(), *sa1b1 = F();
  // t1 (cin=128, dim=64)
  const float *t1sw = F(), *t1sb = F(), *t1ew = F(), *t1eb = F();
  const float *t1qw = F(), *t1qb = F(), *t1kw = F(), *t1kb = F(), *t1vw = F(), *t1vb = F();
  const float *t1p1w = F(), *t1p1b = F(), *t1p2w = F(), *t1p2b = F();
  const float *t1a1w = F(), *t1a1b = F(), *t1a2w = F(), *t1a2b = F();
  const float *sa2w0 = F(), *sa2b0 = F(), *sa2w1 = F(), *sa2b1 = F();
  // t2 (cin=256, dim=64)
  const float *t2sw = F(), *t2sb = F(), *t2ew = F(), *t2eb = F();
  const float *t2qw = F(), *t2qb = F(), *t2kw = F(), *t2kb = F(), *t2vw = F(), *t2vb = F();
  const float *t2p1w = F(), *t2p1b = F(), *t2p2w = F(), *t2p2b = F();
  const float *t2a1w = F(), *t2a1b = F(), *t2a2w = F(), *t2a2b = F();
  const float *sa4w0 = F(), *sa4b0 = F(), *sa4w1 = F(), *sa4b1 = F();
  const float *ps0w = F(), *ps0b = F();
  const float *m1c1w = F(), *m1c1b = F(), *m1c2w = F(), *m1c2b = F(), *m1scw = F(), *m1scb = F();
  const float *ps1w = F(), *ps1b = F();
  const float *ps2w = F(), *ps2b = F();
  const float *r1dict = F(), *r1qw = F(), *r1qb = F(), *r1kw = F(), *r1kb = F();
  const float *r2dict = F(), *r2qw = F(), *r2qb = F(), *r2kw = F(), *r2kb = F();
  const float *m3c1w = F(), *m3c1b = F(), *m3c2w = F(), *m3c2b = F(), *m3scw = F(), *m3scb = F();

  float* dout = (float*)d_out;
  float* l4     = dout;                 // (B,384,1)     [3072]
  float* o_xyz  = dout + 3072;          // (B,3,1024)    [24576]
  float* o_fea  = dout + 27648;         // (B,128,1024)  [1048576]
  float* o_cns1 = dout + 1076224;       // (256,256)
  float* o_cns2 = dout + 1141760;       // (128,128)

  // --- workspace bump allocator (floats) ---
  float* W0 = (float*)d_ws;
  size_t cur = 0;
  auto alloc = [&](size_t n) { float* p = W0 + cur; cur += (n + 63) & ~(size_t)63; return p; };

  // persistent cross-stage tensors
  float* l1_xyz  = alloc(8 * 3 * 512);
  float* l1_feaA = alloc(8 * 128 * 512);   // pre-transformer
  float* l1_feaT = alloc(8 * 128 * 512);   // post-transformer
  float* l2_xyz  = alloc(8 * 3 * 128);
  float* l2_feaA = alloc(8 * 256 * 128);
  float* l2_feaT = alloc(8 * 256 * 128);
  const size_t mark = cur;

  // ======================= SA1: 2048 -> 512, KNN 16 =======================
  {
    cur = mark;
    float* fpsd = alloc(8 * 2048);
    int*   fidx = (int*)alloc(8 * 512);
    int*   idx  = (int*)alloc(8 * 512 * 16);
    float* g1   = alloc((size_t)8 * 6 * 512 * 16);
    float* h1a  = alloc((size_t)8 * 32 * 8192);
    float* h1b  = alloc((size_t)8 * 128 * 8192);

    fps_kernel<<<B, 256, 0, st>>>(pc, fidx, fpsd, 2048, 512);
    gather_cn_kernel<<<grid1(8L * 3 * 512), 256, 0, st>>>(l1_xyz, pc, fidx, B, 3, 2048, 512);
    knn_kernel<<<dim3(8 * 512 / 8), 256, 0, st>>>(l1_xyz, pc, idx, B, 512, 2048);
    group_kernel<<<grid1(8L * 6 * 512 * 16), 256, 0, st>>>(g1, pc, l1_xyz, pc, idx, B, 3, 2048, 512, 16);
    gemm(st, B, 32, 8192, 6, sa1w0, 0, 6, 1, g1, 6L * 8192, 8192, 1, sa1b0,
         h1a, 32L * 8192, 8192, 1.f, 1);
    gemm(st, B, 128, 8192, 32, sa1w1, 0, 32, 1, h1a, 32L * 8192, 8192, 1, sa1b1,
         h1b, 128L * 8192, 8192);
    maxpool_last_kernel<<<grid1(8L * 128 * 512), 256, 0, st>>>(l1_feaA, h1b, 8L * 128 * 512, 16);
  }

  // ======================= Transformer 1 (N=512, dim=64) ==================
  {
    cur = mark;
    const int N = 512;
    float* x1  = alloc((size_t)8 * 64 * N);
    float* kf  = alloc((size_t)8 * 64 * N);
    float* qf  = alloc((size_t)8 * 64 * N);
    float* vf  = alloc((size_t)8 * 64 * N);
    float* agg = alloc((size_t)8 * 64 * N);
    int*   idx = (int*)alloc((size_t)8 * N * 16);
    float* QK  = alloc((size_t)8 * 64 * N * 16);
    float* PR  = alloc((size_t)8 * 3 * N * 16);
    float* PE  = alloc((size_t)8 * 64 * N * 16);
    float* ATT = alloc((size_t)8 * 64 * N * 16);
    float* HID = alloc((size_t)8 * 256 * 2048);   // chunked attn hidden

    long NK = (long)N * 16;
    gemm(st, B, 64, N, 128, t1sw, 0, 128, 1, l1_feaA, 128L * N, N, 1, t1sb, x1, 64L * N, N);
    knn_kernel<<<dim3(8 * N / 8), 256, 0, st>>>(l1_xyz, l1_xyz, idx, B, N, N);
    gemm(st, B, 64, N, 64, t1kw, 0, 64, 1, x1, 64L * N, N, 1, t1kb, kf, 64L * N, N);
    gemm(st, B, 64, N, 64, t1qw, 0, 64, 1, x1, 64L * N, N, 1, t1qb, qf, 64L * N, N);
    gemm(st, B, 64, N, 64, t1vw, 0, 64, 1, x1, 64L * N, N, 1, t1vb, vf, 64L * N, N);
    bsg_kernel<<<grid1(8L * 64 * NK), 256, 0, st>>>(QK, qf, kf, idx, B, 64, N, 16);
    bsg_kernel<<<grid1(8L * 3 * NK), 256, 0, st>>>(PR, l1_xyz, l1_xyz, idx, B, 3, N, 16);
    // pos_emb = pos2(relu(pos1(PR)))   (reuse ATT as hidden-64 scratch)
    gemm(st, B, 64, (int)NK, 3, t1p1w, 0, 3, 1, PR, 3L * NK, NK, 1, t1p1b, ATT, 64L * NK, NK, 1.f, 1);
    gemm(st, B, 64, (int)NK, 64, t1p2w, 0, 64, 1, ATT, 64L * NK, NK, 1, t1p2b, PE, 64L * NK, NK);
    add_inplace_kernel<<<grid1(8L * 64 * NK), 256, 0, st>>>(QK, PE, 8L * 64 * NK);
    // attn = attn2(relu(attn1(QK))) in 2048-wide chunks
    for (long c0 = 0; c0 < NK; c0 += 2048) {
      gemm(st, B, 256, 2048, 64, t1a1w, 0, 64, 1, QK + c0, 64L * NK, NK, 1, t1a1b,
           HID, 256L * 2048, 2048, 1.f, 1);
      gemm(st, B, 64, 2048, 256, t1a2w, 0, 256, 1, HID, 256L * 2048, 2048, 1, t1a2b,
           ATT + c0, 64L * NK, NK);
    }
    softmax_last_kernel<<<grid1(8L * 64 * N), 256, 0, st>>>(ATT, 8L * 64 * N, 16);
    attn_agg_kernel<<<grid1(8L * 64 * N), 256, 0, st>>>(agg, ATT, vf, PE, B, 64, N, 16);
    gemm(st, B, 128, N, 64, t1ew, 0, 64, 1, agg, 64L * N, N, 1, t1eb,
         l1_feaT, 128L * N, N, 1.f, 0, l1_feaA, 128L * N, N);
  }

  // ======================= SA2: 512 -> 128, KNN 16 ========================
  {
    cur = mark;
    float* fpsd = alloc(8 * 512);
    int*   fidx = (int*)alloc(8 * 128);
    int*   idx  = (int*)alloc(8 * 128 * 16);
    float* g2   = alloc((size_t)8 * 131 * 128 * 16);
    float* h2a  = alloc((size_t)8 * 128 * 2048);
    float* h2b  = alloc((size_t)8 * 256 * 2048);

    fps_kernel<<<B, 256, 0, st>>>(l1_xyz, fidx, fpsd, 512, 128);
    gather_cn_kernel<<<grid1(8L * 3 * 128), 256, 0, st>>>(l2_xyz, l1_xyz, fidx, B, 3, 512, 128);
    knn_kernel<<<dim3(8 * 128 / 8), 256, 0, st>>>(l2_xyz, l1_xyz, idx, B, 128, 512);
    group_kernel<<<grid1(8L * 131 * 128 * 16), 256, 0, st>>>(g2, l1_xyz, l2_xyz, l1_feaT, idx,
                                                             B, 128, 512, 128, 16);
    gemm(st, B, 128, 2048, 131, sa2w0, 0, 131, 1, g2, 131L * 2048, 2048, 1, sa2b0,
         h2a, 128L * 2048, 2048, 1.f, 1);
    gemm(st, B, 256, 2048, 128, sa2w1, 0, 128, 1, h2a, 128L * 2048, 2048, 1, sa2b1,
         h2b, 256L * 2048, 2048);
    maxpool_last_kernel<<<grid1(8L * 256 * 128), 256, 0, st>>>(l2_feaA, h2b, 8L * 256 * 128, 16);
  }

  // ======================= Transformer 2 (N=128, dim=64) ==================
  {
    cur = mark;
    const int N = 128;
    float* x1  = alloc((size_t)8 * 64 * N);
    float* kf  = alloc((size_t)8 * 64 * N);
    float* qf  = alloc((size_t)8 * 64 * N);
    float* vf  = alloc((size_t)8 * 64 * N);
    float* agg = alloc((size_t)8 * 64 * N);
    int*   idx = (int*)alloc((size_t)8 * N * 16);
    float* QK  = alloc((size_t)8 * 64 * N * 16);
    float* PR  = alloc((size_t)8 * 3 * N * 16);
    float* PE  = alloc((size_t)8 * 64 * N * 16);
    float* ATT = alloc((size_t)8 * 64 * N * 16);
    float* HID = alloc((size_t)8 * 256 * N * 16);

    long NK = (long)N * 16;
    gemm(st, B, 64, N, 256, t2sw, 0, 256, 1, l2_feaA, 256L * N, N, 1, t2sb, x1, 64L * N, N);
    knn_kernel<<<dim3(8 * N / 8), 256, 0, st>>>(l2_xyz, l2_xyz, idx, B, N, N);
    gemm(st, B, 64, N, 64, t2kw, 0, 64, 1, x1, 64L * N, N, 1, t2kb, kf, 64L * N, N);
    gemm(st, B, 64, N, 64, t2qw, 0, 64, 1, x1, 64L * N, N, 1, t2qb, qf, 64L * N, N);
    gemm(st, B, 64, N, 64, t2vw, 0, 64, 1, x1, 64L * N, N, 1, t2vb, vf, 64L * N, N);
    bsg_kernel<<<grid1(8L * 64 * NK), 256, 0, st>>>(QK, qf, kf, idx, B, 64, N, 16);
    bsg_kernel<<<grid1(8L * 3 * NK), 256, 0, st>>>(PR, l2_xyz, l2_xyz, idx, B, 3, N, 16);
    gemm(st, B, 64, (int)NK, 3, t2p1w, 0, 3, 1, PR, 3L * NK, NK, 1, t2p1b, ATT, 64L * NK, NK, 1.f, 1);
    gemm(st, B, 64, (int)NK, 64, t2p2w, 0, 64, 1, ATT, 64L * NK, NK, 1, t2p2b, PE, 64L * NK, NK);
    add_inplace_kernel<<<grid1(8L * 64 * NK), 256, 0, st>>>(QK, PE, 8L * 64 * NK);
    gemm(st, B, 256, (int)NK, 64, t2a1w, 0, 64, 1, QK, 64L * NK, NK, 1, t2a1b,
         HID, 256L * NK, NK, 1.f, 1);
    gemm(st, B, 64, (int)NK, 256, t2a2w, 0, 256, 1, HID, 256L * NK, NK, 1, t2a2b,
         ATT, 64L * NK, NK);
    softmax_last_kernel<<<grid1(8L * 64 * N), 256, 0, st>>>(ATT, 8L * 64 * N, 16);
    attn_agg_kernel<<<grid1(8L * 64 * N), 256, 0, st>>>(agg, ATT, vf, PE, B, 64, N, 16);
    gemm(st, B, 256, N, 64, t2ew, 0, 64, 1, agg, 64L * N, N, 1, t2eb,
         l2_feaT, 256L * N, N, 1.f, 0, l2_feaA, 256L * N, N);
  }

  // ======================= SA4 (group all, N=128) =========================
  {
    cur = mark;
    float* g4  = alloc((size_t)8 * 259 * 128);
    float* h4a = alloc((size_t)8 * 384 * 128);
    float* h4b = alloc((size_t)8 * 384 * 128);
    chan_cat_kernel<<<grid1(8L * 259 * 128), 256, 0, st>>>(g4, l2_xyz, 3, l2_feaT, 256, 128, 0, B);
    gemm(st, B, 384, 128, 259, sa4w0, 0, 259, 1, g4, 259L * 128, 128, 1, sa4b0,
         h4a, 384L * 128, 128, 1.f, 1);
    gemm(st, B, 384, 128, 384, sa4w1, 0, 384, 1, h4a, 384L * 128, 128, 1, sa4b1,
         h4b, 384L * 128, 128);
    maxpool_last_kernel<<<grid1(8L * 384), 256, 0, st>>>(l4, h4b, 8L * 384, 128);
  }

  // ======================= Decoder ========================================
  {
    cur = mark;
    float* u0a  = alloc((size_t)8 * 256 * 64);
    float* cat1 = alloc((size_t)8 * 640 * 64);
    float* mh   = alloc((size_t)8 * 256 * 64);
    float* msc  = alloc((size_t)8 * 256 * 64);
    float* u0   = alloc((size_t)8 * 256 * 64);
    float* u1a  = alloc((size_t)8 * 256 * 128);
    float* fm1  = alloc(256 * 256);
    float* q1   = alloc((size_t)8 * 64 * 128);
    float* k1   = alloc((size_t)8 * 64 * 256);
    float* SC1  = alloc((size_t)8 * 128 * 256);
    float* O1   = alloc((size_t)8 * 256 * 128);
    float* u1   = alloc((size_t)8 * 256 * 128);
    float* u1c  = alloc((size_t)8 * 256 * 256);
    float* u2a  = alloc((size_t)8 * 128 * 512);
    float* fm2  = alloc(128 * 128);
    float* q2   = alloc((size_t)8 * 64 * 512);
    float* k2   = alloc((size_t)8 * 64 * 128);
    float* SC2  = alloc((size_t)8 * 512 * 128);
    float* O2   = alloc((size_t)8 * 128 * 512);
    float* u2   = alloc((size_t)8 * 128 * 512);
    float* mh3  = alloc((size_t)8 * 32 * 512);
    float* sc3  = alloc((size_t)8 * 3 * 512);
    float* xyz2 = alloc((size_t)8 * 3 * 512);

    // ps0: (B,384,1) -> (B,256,64)
    tconv_kernel<<<grid1(8L * 256 * 64), 256, 0, st>>>(l4, ps0w, ps0b, u0a, B, 384, 256, 1, 64);
    // mlp1 on concat(u0a, broadcast l4)
    chan_cat_kernel<<<grid1(8L * 640 * 64), 256, 0, st>>>(cat1, u0a, 256, l4, 384, 64, 1, B);
    gemm(st, B, 256, 64, 640, m1c1w, 0, 640, 1, cat1, 640L * 64, 64, 1, m1c1b, mh, 256L * 64, 64, 1.f, 1);
    gemm(st, B, 256, 64, 640, m1scw, 0, 640, 1, cat1, 640L * 64, 64, 1, m1scb, msc, 256L * 64, 64);
    gemm(st, B, 256, 64, 256, m1c2w, 0, 256, 1, mh, 256L * 64, 64, 1, m1c2b,
         u0, 256L * 64, 64, 1.f, 0, msc, 256L * 64, 64);
    // ps1: (B,256,64) -> (B,256,128)
    tconv_kernel<<<grid1(8L * 256 * 128), 256, 0, st>>>(u0, ps1w, ps1b, u1a, B, 256, 256, 64, 2);
    // fea_refine 1 (dim=256, nd=256, Nf=128)
    colnorm_kernel<<<grid1(256), 256, 0, st>>>(r1dict, fm1, 256, 256);
    gemm(st, 1, 256, 256, 256, fm1, 0, 1, 256, fm1, 0, 256, 1, nullptr, o_cns1, 0, 256);
    gemm(st, B, 64, 128, 256, r1qw, 0, 256, 1, u1a, 256L * 128, 128, 1, r1qb, q1, 64L * 128, 128);
    gemm(st, B, 64, 256, 256, r1kw, 0, 256, 1, r1dict, 0, 256, 1, r1kb, k1, 64L * 256, 256);
    gemm(st, B, 128, 256, 64, q1, 64L * 128, 1, 128, k1, 64L * 256, 256, 1, nullptr,
         SC1, 128L * 256, 256, 0.125f);
    softmax_last_kernel<<<grid1(8L * 128), 256, 0, st>>>(SC1, 8L * 128, 256);
    gemm(st, B, 256, 128, 256, r1dict, 0, 256, 1, SC1, 128L * 256, 1, 256, nullptr,
         O1, 256L * 128, 128);
    avg2_kernel<<<grid1(8L * 256 * 128), 256, 0, st>>>(u1, O1, u1a, 8L * 256 * 128);
    // concat(l2_feaT, u1) along points
    copy_into_kernel<<<grid1(8L * 256 * 128), 256, 0, st>>>(u1c, 256, 0, l2_feaT, 128, 8L * 256);
    copy_into_kernel<<<grid1(8L * 256 * 128), 256, 0, st>>>(u1c, 256, 128, u1, 128, 8L * 256);
    // ps2: (B,256,256) -> (B,128,512)
    tconv_kernel<<<grid1(8L * 128 * 512), 256, 0, st>>>(u1c, ps2w, ps2b, u2a, B, 256, 128, 256, 2);
    // fea_refine 2 (dim=128, nd=128, Nf=512)
    colnorm_kernel<<<grid1(128), 256, 0, st>>>(r2dict, fm2, 128, 128);
    gemm(st, 1, 128, 128, 128, fm2, 0, 1, 128, fm2, 0, 128, 1, nullptr, o_cns2, 0, 128);
    gemm(st, B, 64, 512, 128, r2qw, 0, 128, 1, u2a, 128L * 512, 512, 1, r2qb, q2, 64L * 512, 512);
    gemm(st, B, 64, 128, 128, r2kw, 0, 128, 1, r2dict, 0, 128, 1, r2kb, k2, 64L * 128, 128);
    gemm(st, B, 512, 128, 64, q2, 64L * 512, 1, 512, k2, 64L * 128, 128, 1, nullptr,
         SC2, 512L * 128, 128, 0.125f);
    softmax_last_kernel<<<grid1(8L * 512), 256, 0, st>>>(SC2, 8L * 512, 128);
    gemm(st, B, 128, 512, 128, r2dict, 0, 128, 1, SC2, 512L * 128, 1, 128, nullptr,
         O2, 128L * 512, 512);
    avg2_kernel<<<grid1(8L * 128 * 512), 256, 0, st>>>(u2, O2, u2a, 8L * 128 * 512);
    // mlp3 -> refined xyz (B,3,512)
    gemm(st, B, 32, 512, 128, m3c1w, 0, 128, 1, u2, 128L * 512, 512, 1, m3c1b,
         mh3, 32L * 512, 512, 1.f, 1);
    gemm(st, B, 3, 512, 128, m3scw, 0, 128, 1, u2, 128L * 512, 512, 1, m3scb,
         sc3, 3L * 512, 512);
    gemm(st, B, 3, 512, 32, m3c2w, 0, 32, 1, mh3, 32L * 512, 512, 1, m3c2b,
         xyz2, 3L * 512, 512, 1.f, 0, sc3, 3L * 512, 512);
    // final concats into d_out
    copy_into_kernel<<<grid1(8L * 3 * 512), 256, 0, st>>>(o_xyz, 1024, 0,   l1_xyz, 512, 8L * 3);
    copy_into_kernel<<<grid1(8L * 3 * 512), 256, 0, st>>>(o_xyz, 1024, 512, xyz2,   512, 8L * 3);
    copy_into_kernel<<<grid1(8L * 128 * 512), 256, 0, st>>>(o_fea, 1024, 0,   l1_feaT, 512, 8L * 128);
    copy_into_kernel<<<grid1(8L * 128 * 512), 256, 0, st>>>(o_fea, 1024, 512, u2,     512, 8L * 128);
  }
}
